// ConvLRULayer_84791244357925
// MI455X (gfx1250) — matrix operand order repaired and checked by reading, not compiled, not run on hardware
//
// ConvLRULayer for MI455X (gfx1250, wave32, WMMA + TDM).
//
// Math collapse used (exact, up to fp rounding):
//  * ortho DFT is unitary+symmetric  =>  IFFT2(proj(FFT2(xp))) = proj(xp);
//    the frequency-domain bias pb becomes a delta at padded row s=0 which the
//    crop [pad:-pad] deletes  =>  hs[b,l,d,s,w] = sum_c x[b,l,c,s,w] * pW[c,d].
//  * zq = U^H FFT2(hs) V = (F^H U)^H hs (F V) = Ahat^H hs Bhat   per (b,l,c)
//  * h_sp = IFFT2(U z V^H) = Ahat z Bhat^H                       per (b,l,c)
//  * conv_r/conv_i + 1x1 projc merge into one 3x3 conv with weights
//    Wmr[d,c,t] = sum_e projc[d,e] convr_w[e,c,t] (and Wmi with projc[d,C+e]).
// All GEMMs run on v_wmma_f32_16x16x32_bf16 (bf16 in, f32 accumulate).
// hs tiles are staged into LDS with the Tensor Data Mover (tensor_load_to_lds).

#include <hip/hip_runtime.h>

#define Bb 2
#define Ll 32
#define Cc 32
#define Ss 64
#define Ww 64
#define Rr 32
#define MHh 32
#define SP (Ss * Ww)   // 4096

typedef __bf16 bf16_t;
typedef __attribute__((ext_vector_type(16))) __bf16 bf16x16;
typedef __attribute__((ext_vector_type(8)))  float  f32x8;
typedef __attribute__((ext_vector_type(4)))  unsigned int u32x4;
typedef __attribute__((ext_vector_type(8)))  unsigned int u32x8;

// ---------------------------------------------------------------- WMMA helpers

__device__ __forceinline__ f32x8 zero8() {
  f32x8 z = {0.f, 0.f, 0.f, 0.f, 0.f, 0.f, 0.f, 0.f};
  return z;
}

__device__ __forceinline__ f32x8 wmma_bf16(bf16x16 a, bf16x16 b, f32x8 c) {
  return __builtin_amdgcn_wmma_f32_16x16x32_bf16(
      /*neg_a=*/false, a, /*neg_b=*/false, b,
      /*c_mod=*/(short)0, c, /*reuse_a=*/false, /*reuse_b=*/false);
}

// A-matrix fragment (16x32, row-major source A[m][k], leading dim ld).
// Lanes 0-15: elems 0..15 -> K=0..15.
// Lanes16-31: elems 0..15 -> K=16..31.
__device__ __forceinline__ bf16x16 fragA(const bf16_t* A, int ld, int m0, int k0) {
  int lane = threadIdx.x & 31;
  int half = lane >> 4;
  int m = m0 + (lane & 15);
  bf16x16 f;
#pragma unroll
  for (int e = 0; e < 16; ++e) {
    int k = k0 + e + 16 * half;
    f[e] = A[m * ld + k];
  }
  return f;
}

// B-matrix fragment (32x16, row-major source B[k][n], leading dim ld).
// Lanes 0-15 hold K=0..15, lanes 16-31 hold K=16..31; column n = lane%16.
__device__ __forceinline__ bf16x16 fragB(const bf16_t* Bm, int ld, int k0, int n0) {
  int lane = threadIdx.x & 31;
  int half = lane >> 4;
  int n = n0 + (lane & 15);
  bf16x16 f;
#pragma unroll
  for (int e = 0; e < 16; ++e) {
    int k = k0 + e + 16 * half;
    f[e] = Bm[k * ld + n];
  }
  return f;
}

// C/D fragment store: VGPR i -> row m0+i (lanes 0-15) / m0+i+8 (lanes 16-31).
template <typename T>
__device__ __forceinline__ void store_cfrag(T* dst, int ld, int m0, int n0, f32x8 c) {
  int lane = threadIdx.x & 31;
  int half = lane >> 4;
  int n = n0 + (lane & 15);
#pragma unroll
  for (int i = 0; i < 8; ++i) dst[(size_t)(m0 + i + 8 * half) * ld + n] = (T)c[i];
}

// Conv weight fragment: W[(m*Cc + k)*9 + t]   (A-matrix layout, K = input chan).
__device__ __forceinline__ bf16x16 fragW(const bf16_t* Wl, int t, int m0) {
  int lane = threadIdx.x & 31;
  int half = lane >> 4;
  int m = m0 + (lane & 15);
  bf16x16 f;
#pragma unroll
  for (int e = 0; e < 16; ++e) {
    int k = e + 16 * half;
    f[e] = Wl[(m * Cc + k) * 9 + t];
  }
  return f;
}

// Shifted conv input fragment from LDS halo tile xt[c][6][66]; n encodes 4 s-rows x 64 w.
__device__ __forceinline__ bf16x16 fragConvB(const bf16_t* xt, int n0, int dy, int dx) {
  int lane = threadIdx.x & 31;
  int half = lane >> 4;
  int n = n0 + (lane & 15);
  int sl = (n >> 6) + 1 + dy;   // 0..5
  int wc = (n & 63) + 1 + dx;   // 0..65
  bf16x16 f;
#pragma unroll
  for (int e = 0; e < 16; ++e) {
    int k = e + 16 * half;
    f[e] = xt[(k * 6 + sl) * 66 + wc];
  }
  return f;
}

// ----------------------------------------------------- Tensor Data Mover (TDM)
// Async DMA of one contiguous row of n 2-byte elements: global -> LDS.
// D# built per CDNA5 ISA; 2-group form (tensor <= 2D).
// Generic LDS pointers carry the workgroup-relative byte offset in addr[31:0].
__device__ __forceinline__ void tdm_load_row_b16(void* lds_dst, const void* gsrc,
                                                 unsigned n) {
  unsigned long long ga = (unsigned long long)gsrc;
  unsigned lo = (unsigned)(unsigned long long)lds_dst;
  u32x4 g0;
  g0[0] = 1u;                                          // count=1 (valid, user mode)
  g0[1] = lo;                                          // lds_addr (bytes)
  g0[2] = (unsigned)ga;                                // global_addr[31:0]
  g0[3] = ((unsigned)(ga >> 32) & 0x01FFFFFFu)         // global_addr[56:32]
          | (2u << 30);                                // type = 2 ("image")
  u32x8 g1;
  g1[0] = 1u << 16;                                    // wg_mask=0, data_size=1 (2B)
  g1[1] = (n & 0xFFFFu) << 16;                         // tensor_dim0[15:0]
  g1[2] = (n >> 16) | (1u << 16);                      // tensor_dim0[31:16] | tensor_dim1=1
  g1[3] = (n & 0xFFFFu) << 16;                         // tile_dim0
  g1[4] = 1u;                                          // tile_dim1=1, tile_dim2=0
  g1[5] = n;                                           // tensor_dim0_stride[31:0]
  g1[6] = 0u;                                          // stride hi / dim1_stride lo
  g1[7] = 0u;
  asm volatile("tensor_load_to_lds %0, %1" :: "s"(g0), "s"(g1) : "memory");
}

// ------------------------------------------------------------------- kernels

// Fold ortho-DFT into the spectral bases: Ahat = F^H U, Bhat = F V  ([C,64,R]).
__global__ void __launch_bounds__(256)
k_precompute_AB(const float* U_re, const float* U_im,
                const float* V_re, const float* V_im,
                float* Ahat_re, float* Ahat_im, float* Bhat_re, float* Bhat_im) {
  int idx = blockIdx.x * blockDim.x + threadIdx.x;  // over C*64*R
  if (idx >= Cc * 64 * Rr) return;
  int r = idx & (Rr - 1);
  int s = (idx / Rr) & 63;
  int c = idx / (Rr * 64);
  float are = 0.f, aim = 0.f, bre = 0.f, bim = 0.f;
  for (int t = 0; t < 64; ++t) {
    float th = (6.283185307179586f / 64.f) * (float)((s * t) & 63);
    float sth, cth;
    __sincosf(th, &sth, &cth);
    float ur = U_re[(c * 64 + t) * Rr + r], ui = U_im[(c * 64 + t) * Rr + r];
    are += cth * ur - sth * ui;            // conj(F) = e^{+i th}
    aim += sth * ur + cth * ui;
    float vr = V_re[(c * 64 + t) * Rr + r], vi = V_im[(c * 64 + t) * Rr + r];
    bre += cth * vr + sth * vi;            // F = e^{-i th}
    bim += cth * vi - sth * vr;
  }
  const float inv = 0.125f;                // 1/sqrt(64)
  Ahat_re[idx] = are * inv; Ahat_im[idx] = aim * inv;
  Bhat_re[idx] = bre * inv; Bhat_im[idx] = bim * inv;
}

// Merge convr/convi with the 1x1 projc into one 3x3, 2-plane conv.
__global__ void __launch_bounds__(256)
k_merge_weights(const float* convr_w, const float* convr_b,
                const float* convi_w, const float* convi_b,
                const float* projc_w, const float* projc_b,
                bf16_t* Wmr, bf16_t* Wmi, float* bias_m) {
  int idx = blockIdx.x * blockDim.x + threadIdx.x;  // over C*C*9
  if (idx >= Cc * Cc * 9) return;
  int t = idx % 9;
  int c = (idx / 9) % Cc;
  int d = idx / (9 * Cc);
  float ar = 0.f, ai = 0.f;
  for (int e = 0; e < Cc; ++e) {
    ar += projc_w[d * (2 * Cc) + e] * convr_w[(e * Cc + c) * 9 + t];
    ai += projc_w[d * (2 * Cc) + Cc + e] * convi_w[(e * Cc + c) * 9 + t];
  }
  Wmr[idx] = (bf16_t)ar;
  Wmi[idx] = (bf16_t)ai;
  if (c == 0 && t == 0) {
    float b = projc_b[d];
    for (int e = 0; e < Cc; ++e) {
      b += projc_w[d * (2 * Cc) + e] * convr_b[e];
      b += projc_w[d * (2 * Cc) + Cc + e] * convi_b[e];
    }
    bias_m[d] = b;
  }
}

// ctx[b,l,c] = mean_{s,w} x
__global__ void __launch_bounds__(256)
k_ctx(const float* x, float* ctx) {
  int blc = blockIdx.x;
  const float* p = x + (size_t)blc * SP;
  float s = 0.f;
  for (int i = threadIdx.x; i < SP; i += 256) s += p[i];
  __shared__ float red[256];
  red[threadIdx.x] = s;
  __syncthreads();
  for (int o = 128; o > 0; o >>= 1) {
    if (threadIdx.x < o) red[threadIdx.x] += red[threadIdx.x + o];
    __syncthreads();
  }
  if (threadIdx.x == 0) ctx[blc] = red[0] * (1.f / SP);
}

// Forcing MLP -> lamb (complex) and gamma, per (b,l,c,r). One block per (b,l).
__global__ void __launch_bounds__(256)
k_forcing(const float* ctx, const float* listT, const float* plb, const float* disp,
          const float* fm_W1, const float* fm_b1, const float* fm_W2, const float* fm_b2,
          const float* f_scale, float* lamb_re, float* lamb_im, float* gammav) {
  int bl = blockIdx.x;
  float dt = listT[bl];
  __shared__ float inp[Cc + 1];
  __shared__ float mid[MHh];
  if (threadIdx.x < Cc) inp[threadIdx.x] = ctx[bl * Cc + threadIdx.x];
  if (threadIdx.x == Cc) inp[Cc] = dt;
  __syncthreads();
  if (threadIdx.x < MHh) {
    float a = fm_b1[threadIdx.x];
    for (int i = 0; i <= Cc; ++i) a += inp[i] * fm_W1[i * MHh + threadIdx.x];
    mid[threadIdx.x] = tanhf(a);
  }
  __syncthreads();
  float fs = f_scale[0];
  for (int cr = threadIdx.x; cr < Cc * Rr; cr += 256) {
    int o0 = cr * 2, o1 = cr * 2 + 1;
    float m0 = fm_b2[o0], m1 = fm_b2[o1];
    for (int h = 0; h < MHh; ++h) {
      m0 += mid[h] * fm_W2[h * (Cc * Rr * 2) + o0];
      m1 += mid[h] * fm_W2[h * (Cc * Rr * 2) + o1];
    }
    float dnu = fs * tanhf(m0), dth = fs * tanhf(m1);
    float nub = __expf(plb[cr] + disp[cr]);
    float thb = __expf(plb[Cc * Rr + cr] + disp[Cc * Rr + cr]);
    float nu_t = fmaxf(nub * dt + dnu, 1e-6f);
    float th_t = thb * dt + dth;
    float el = __expf(-nu_t);
    float sn, cs;
    __sincosf(th_t, &sn, &cs);
    size_t o = (size_t)bl * Cc * Rr + cr;
    lamb_re[o] = el * cs;
    lamb_im[o] = el * sn;
    gammav[o] = sqrtf(fmaxf(1.f - __expf(-2.f * nu_t), 1e-12f));
  }
}

// hs[d,n] = sum_c pW[c,d] * x[c,n]   (complex out, real in). WMMA M32/K32/N128 per WG.
__global__ void __launch_bounds__(256)
k_proj(const float* x, const float* pW_re, const float* pW_im,
       bf16_t* hs_re, bf16_t* hs_im) {
  int bl = blockIdx.x >> 5;
  int n0 = (blockIdx.x & 31) * 128;
  __shared__ bf16_t pwr[32 * 32], pwi[32 * 32], xs[32 * 128];
  int tid = threadIdx.x;
  for (int i = tid; i < 1024; i += 256) {
    int d = i >> 5, cin = i & 31;
    pwr[i] = (bf16_t)pW_re[cin * Cc + d];   // A[d][c] = pW[c][d]
    pwi[i] = (bf16_t)pW_im[cin * Cc + d];
  }
  for (int i = tid; i < 32 * 128; i += 256) {
    int cin = i >> 7, n = i & 127;
    xs[i] = (bf16_t)x[((size_t)bl * Cc + cin) * SP + n0 + n];
  }
  __syncthreads();
  int wave = tid >> 5;
  int nsub = wave * 16;
  size_t obase = (size_t)bl * Cc * SP + n0 + nsub;
#pragma unroll
  for (int mt = 0; mt < 2; ++mt) {
    bf16x16 aR = fragA(pwr, 32, mt * 16, 0);
    bf16x16 aI = fragA(pwi, 32, mt * 16, 0);
    bf16x16 bX = fragB(xs, 128, 0, nsub);
    f32x8 accR = wmma_bf16(aR, bX, zero8());
    f32x8 accI = wmma_bf16(aI, bX, zero8());
    store_cfrag(hs_re + obase, SP, mt * 16, 0, accR);
    store_cfrag(hs_im + obase, SP, mt * 16, 0, accI);
  }
}

// zq = Ahat^H hs Bhat per (b,l,c): complex (32x64)(64x64)(64x32) via two WMMA GEMMs.
// hs tile (2 x 8KB, contiguous bf16) staged via the Tensor Data Mover.
__global__ void __launch_bounds__(256)
k_zq(const bf16_t* hs_re, const bf16_t* hs_im,
     const float* Ahat_re, const float* Ahat_im,
     const float* Bhat_re, const float* Bhat_im,
     float* zq_re, float* zq_im) {
  int bl = blockIdx.x >> 5, c = blockIdx.x & 31;
  __shared__ bf16_t hRe[64 * 64], hIm[64 * 64];
  __shared__ bf16_t aRe[32 * 64], aImP[32 * 64], aImN[32 * 64];
  __shared__ bf16_t bRe[64 * 32], bIm[64 * 32], bImN[64 * 32];
  __shared__ bf16_t tRe[32 * 64], tIm[32 * 64];
  int tid = threadIdx.x;
  size_t hbase = ((size_t)bl * Cc + c) * SP;
  if (tid < 32) {  // wave 0 drives the TDM; EXEC is ignored by tensor ops
    tdm_load_row_b16(hRe, hs_re + hbase, SP);
    tdm_load_row_b16(hIm, hs_im + hbase, SP);
    __builtin_amdgcn_s_wait_tensorcnt(0);
  }
  for (int i = tid; i < 32 * 64; i += 256) {   // A' = conj(Ahat)^T : [r][s]
    int r = i >> 6, s = i & 63;
    float vre = Ahat_re[(c * 64 + s) * Rr + r];
    float vim = Ahat_im[(c * 64 + s) * Rr + r];
    aRe[i] = (bf16_t)vre; aImP[i] = (bf16_t)vim; aImN[i] = (bf16_t)(-vim);
  }
  for (int i = tid; i < 64 * 32; i += 256) {   // B = Bhat : [w][q]
    int w = i >> 5, q = i & 31;
    float vre = Bhat_re[(c * 64 + w) * Rr + q];
    float vim = Bhat_im[(c * 64 + w) * Rr + q];
    bRe[i] = (bf16_t)vre; bIm[i] = (bf16_t)vim; bImN[i] = (bf16_t)(-vim);
  }
  __syncthreads();
  int wave = tid >> 5;
  {  // T = A'^ * hs   (32x64), complex: Tre = Are*Hre + Aim*Him ; Tim = Are*Him - Aim*Hre
    int m0 = (wave >> 2) * 16, n0 = (wave & 3) * 16;
    f32x8 accR = zero8(), accI = zero8();
#pragma unroll
    for (int kc = 0; kc < 64; kc += 32) {
      bf16x16 fr = fragA(aRe, 64, m0, kc);
      bf16x16 fp = fragA(aImP, 64, m0, kc);
      bf16x16 fn = fragA(aImN, 64, m0, kc);
      bf16x16 hR = fragB(hRe, 64, kc, n0);
      bf16x16 hI = fragB(hIm, 64, kc, n0);
      accR = wmma_bf16(fr, hR, accR);
      accR = wmma_bf16(fp, hI, accR);   // +Aim*Him  (conj handled via sign staging)
      accI = wmma_bf16(fr, hI, accI);
      accI = wmma_bf16(fn, hR, accI);   // -Aim*Hre
    }
    store_cfrag(tRe, 64, m0, n0, accR);
    store_cfrag(tIm, 64, m0, n0, accI);
  }
  __syncthreads();
  {  // zq = T * B   (32x32): zre = Tre*Bre - Tim*Bim ; zim = Tre*Bim + Tim*Bre
    int part = wave >> 2;
    int m0 = ((wave >> 1) & 1) * 16, n0 = (wave & 1) * 16;
    f32x8 acc = zero8();
#pragma unroll
    for (int kc = 0; kc < 64; kc += 32) {
      bf16x16 fTr = fragA(tRe, 64, m0, kc);
      bf16x16 fTi = fragA(tIm, 64, m0, kc);
      if (part == 0) {
        acc = wmma_bf16(fTr, fragB(bRe, 32, kc, n0), acc);
        acc = wmma_bf16(fTi, fragB(bImN, 32, kc, n0), acc);
      } else {
        acc = wmma_bf16(fTr, fragB(bIm, 32, kc, n0), acc);
        acc = wmma_bf16(fTi, fragB(bRe, 32, kc, n0), acc);
      }
    }
    size_t zbase = ((size_t)bl * Cc + c) * (Rr * Rr);
    if (part == 0) store_cfrag(zq_re + zbase, Rr, m0, n0, acc);
    else           store_cfrag(zq_im + zbase, Rr, m0, n0, acc);
  }
}

// Linear recurrence over L (in place on zq): z_l = lamb_l * z_{l-1} + gamma_l * zq_l.
__global__ void __launch_bounds__(256)
k_scan(float* zr, float* zi, const float* lre, const float* lim, const float* gam) {
  int idx = blockIdx.x * blockDim.x + threadIdx.x;  // over B * C*R*R
  if (idx >= Bb * Cc * Rr * Rr) return;
  int b = idx / (Cc * Rr * Rr);
  int crq = idx - b * (Cc * Rr * Rr);
  int c = crq / (Rr * Rr);
  int r = (crq / Rr) & (Rr - 1);
  float zre = 0.f, zim = 0.f;
  for (int l = 0; l < Ll; ++l) {
    size_t zo = ((size_t)(b * Ll + l) * Cc) * (Rr * Rr) + crq;
    size_t lo = (size_t)(b * Ll + l) * Cc * Rr + c * Rr + r;
    float g = gam[lo];
    float xr = zr[zo] * g, xi = zi[zo] * g;
    float ar = lre[lo], ai = lim[lo];
    float nr = ar * zre - ai * zim + xr;
    float ni = ar * zim + ai * zre + xi;
    zre = nr; zim = ni;
    zr[zo] = zre; zi[zo] = zim;
  }
}

// h_sp = Ahat z Bhat^H per (b,l,c): (64x32)(32x32)(32x64), two WMMA GEMMs.
__global__ void __launch_bounds__(256)
k_recon(const float* zq_re, const float* zq_im,
        const float* Ahat_re, const float* Ahat_im,
        const float* Bhat_re, const float* Bhat_im,
        bf16_t* hsp_re, bf16_t* hsp_im) {
  int bl = blockIdx.x >> 5, c = blockIdx.x & 31;
  __shared__ bf16_t zRe[32 * 32], zIm[32 * 32];
  __shared__ bf16_t aRe[64 * 32], aImP[64 * 32], aImN[64 * 32];
  __shared__ bf16_t qRe[32 * 64], qImP[32 * 64], qImN[32 * 64];
  __shared__ bf16_t pRe[64 * 32], pIm[64 * 32];
  int tid = threadIdx.x;
  size_t zbase = ((size_t)bl * Cc + c) * (Rr * Rr);
  for (int i = tid; i < 32 * 32; i += 256) {
    zRe[i] = (bf16_t)zq_re[zbase + i];
    zIm[i] = (bf16_t)zq_im[zbase + i];
  }
  for (int i = tid; i < 64 * 32; i += 256) {   // A = Ahat : [s][r] (contiguous copy)
    float vre = Ahat_re[(size_t)c * 64 * Rr + i];
    float vim = Ahat_im[(size_t)c * 64 * Rr + i];
    aRe[i] = (bf16_t)vre; aImP[i] = (bf16_t)vim; aImN[i] = (bf16_t)(-vim);
  }
  for (int i = tid; i < 32 * 64; i += 256) {   // Q[q][w] = Bhat[c][w][q] (transposed)
    int q = i >> 6, w = i & 63;
    float vre = Bhat_re[((size_t)c * 64 + w) * Rr + q];
    float vim = Bhat_im[((size_t)c * 64 + w) * Rr + q];
    qRe[i] = (bf16_t)vre; qImP[i] = (bf16_t)vim; qImN[i] = (bf16_t)(-vim);
  }
  __syncthreads();
  int wave = tid >> 5;
  {  // P = A z : Pre = Are*zre - Aim*zim ; Pim = Are*zim + Aim*zre
    int m0 = (wave >> 1) * 16, n0 = (wave & 1) * 16;
    bf16x16 fA = fragA(aRe, 32, m0, 0);
    bf16x16 fP = fragA(aImP, 32, m0, 0);
    bf16x16 fN = fragA(aImN, 32, m0, 0);
    bf16x16 zR = fragB(zRe, 32, 0, n0);
    bf16x16 zI = fragB(zIm, 32, 0, n0);
    f32x8 accR = wmma_bf16(fA, zR, zero8());
    accR = wmma_bf16(fN, zI, accR);
    f32x8 accI = wmma_bf16(fA, zI, zero8());
    accI = wmma_bf16(fP, zR, accI);
    store_cfrag(pRe, 32, m0, n0, accR);
    store_cfrag(pIm, 32, m0, n0, accI);
  }
  __syncthreads();
  size_t hbase = ((size_t)bl * Cc + c) * SP;
  // h_sp = P * B^H, with B^H = (Qre, -QimP):  re = Pre*Qre + Pim*QimP
  //                                            im = -Pre*QimP + Pim*Qre = Pre*QimN + Pim*Qre
  for (int tp = wave; tp < 32; tp += 8) {
    int part = tp >> 4, t = tp & 15;
    int m0 = (t >> 2) * 16, n0 = (t & 3) * 16;
    bf16x16 fPr = fragA(pRe, 32, m0, 0);
    bf16x16 fPi = fragA(pIm, 32, m0, 0);
    f32x8 acc = zero8();
    if (part == 0) {
      acc = wmma_bf16(fPr, fragB(qRe, 64, 0, n0), acc);
      acc = wmma_bf16(fPi, fragB(qImP, 64, 0, n0), acc);
      store_cfrag(hsp_re + hbase, Ww, m0, n0, acc);
    } else {
      acc = wmma_bf16(fPr, fragB(qImN, 64, 0, n0), acc);
      acc = wmma_bf16(fPi, fragB(qRe, 64, 0, n0), acc);
      store_cfrag(hsp_im + hbase, Ww, m0, n0, acc);
    }
  }
}

// Merged 3x3 conv (2 planes) as 18 shifted WMMA GEMMs; 4 s-rows (N=256) per WG.
__global__ void __launch_bounds__(256)
k_conv(const bf16_t* hsp_re, const bf16_t* hsp_im,
       const bf16_t* Wmr, const bf16_t* Wmi, const float* bias_m, float* hf) {
  int bl = blockIdx.x >> 4;
  int s0 = (blockIdx.x & 15) * 4;
  __shared__ bf16_t xr[32 * 6 * 66], xi[32 * 6 * 66];
  int tid = threadIdx.x;
  for (int i = tid; i < 32 * 6 * 66; i += 256) {
    int c = i / 396;
    int rem = i - c * 396;
    int ss = rem / 66, ww = rem - ss * 66;
    int sg = s0 + ss - 1, wg = ww - 1;
    bool ok = (sg >= 0) && (sg < Ss) && (wg >= 0) && (wg < Ww);
    size_t base = ((size_t)bl * Cc + c) * SP;
    xr[i] = ok ? hsp_re[base + sg * Ww + wg] : (bf16_t)0.0f;
    xi[i] = ok ? hsp_im[base + sg * Ww + wg] : (bf16_t)0.0f;
  }
  __syncthreads();
  int wave = tid >> 5;
  int lane = tid & 31, half = lane >> 4;
  for (int tile = wave; tile < 32; tile += 8) {
    int m0 = (tile >> 4) * 16;
    int n0 = (tile & 15) * 16;
    f32x8 acc;
#pragma unroll
    for (int i = 0; i < 8; ++i) acc[i] = bias_m[m0 + i + 8 * half];
#pragma unroll
    for (int t = 0; t < 9; ++t) {
      int dy = t / 3 - 1, dx = t % 3 - 1;
      acc = wmma_bf16(fragW(Wmr, t, m0), fragConvB(xr, n0, dy, dx), acc);
      acc = wmma_bf16(fragW(Wmi, t, m0), fragConvB(xi, n0, dy, dx), acc);
    }
    store_cfrag(hf + (size_t)bl * Cc * SP + (size_t)s0 * Ww, SP, m0, n0, acc);
  }
}

// LayerNorm over (s,w) per (b,l,c) + residual.
__global__ void __launch_bounds__(256)
k_ln(const float* hf, const float* x, const float* ln_w, const float* ln_b, float* out) {
  int blc = blockIdx.x;
  const float* h = hf + (size_t)blc * SP;
  const float* xx = x + (size_t)blc * SP;
  float s = 0.f, s2 = 0.f;
  for (int i = threadIdx.x; i < SP; i += 256) {
    float v = h[i];
    s += v; s2 += v * v;
  }
  __shared__ float r1[256], r2[256];
  r1[threadIdx.x] = s; r2[threadIdx.x] = s2;
  __syncthreads();
  for (int o = 128; o > 0; o >>= 1) {
    if (threadIdx.x < o) { r1[threadIdx.x] += r1[threadIdx.x + o]; r2[threadIdx.x] += r2[threadIdx.x + o]; }
    __syncthreads();
  }
  float mu = r1[0] * (1.f / SP);
  float var = r2[0] * (1.f / SP) - mu * mu;
  float inv = rsqrtf(var + 1e-5f);
  for (int i = threadIdx.x; i < SP; i += 256)
    out[(size_t)blc * SP + i] = xx[i] + (h[i] - mu) * inv * ln_w[i] + ln_b[i];
}

// ------------------------------------------------------------------ launcher

extern "C" void kernel_launch(void* const* d_in, const int* in_sizes, int n_in,
                              void* d_out, int out_size, void* d_ws, size_t ws_size,
                              hipStream_t stream) {
  (void)in_sizes; (void)n_in; (void)out_size; (void)ws_size;
  const float* x       = (const float*)d_in[0];
  const float* listT   = (const float*)d_in[1];
  const float* plb     = (const float*)d_in[2];
  const float* disp    = (const float*)d_in[3];
  const float* fm_W1   = (const float*)d_in[4];
  const float* fm_b1   = (const float*)d_in[5];
  const float* fm_W2   = (const float*)d_in[6];
  const float* fm_b2   = (const float*)d_in[7];
  const float* f_scale = (const float*)d_in[8];
  const float* U_re    = (const float*)d_in[9];
  const float* U_im    = (const float*)d_in[10];
  const float* V_re    = (const float*)d_in[11];
  const float* V_im    = (const float*)d_in[12];
  const float* pW_re   = (const float*)d_in[13];
  const float* pW_im   = (const float*)d_in[14];
  // d_in[15]/d_in[16] (pb_re/pb_im): provably no effect — the frequency-domain DC
  // bias maps to padded spatial row s=0, which the crop [pad:-pad] removes.
  const float* convr_w = (const float*)d_in[17];
  const float* convr_b = (const float*)d_in[18];
  const float* convi_w = (const float*)d_in[19];
  const float* convi_b = (const float*)d_in[20];
  const float* projc_w = (const float*)d_in[21];
  const float* projc_b = (const float*)d_in[22];
  const float* ln_w    = (const float*)d_in[23];
  const float* ln_b    = (const float*)d_in[24];
  float* out = (float*)d_out;

  char* p = (char*)d_ws;
  auto take = [&](size_t bytes) -> char* {
    char* r = p;
    p += (bytes + 255) & ~(size_t)255;
    return r;
  };
  const size_t nBLC = (size_t)Bb * Ll * Cc;
  float*  Ahat_re = (float*)take(sizeof(float) * Cc * 64 * Rr);
  float*  Ahat_im = (float*)take(sizeof(float) * Cc * 64 * Rr);
  float*  Bhat_re = (float*)take(sizeof(float) * Cc * 64 * Rr);
  float*  Bhat_im = (float*)take(sizeof(float) * Cc * 64 * Rr);
  bf16_t* hs_re   = (bf16_t*)take(sizeof(bf16_t) * nBLC * SP);
  bf16_t* hs_im   = (bf16_t*)take(sizeof(bf16_t) * nBLC * SP);
  float*  zq_re   = (float*)take(sizeof(float) * nBLC * Rr * Rr);
  float*  zq_im   = (float*)take(sizeof(float) * nBLC * Rr * Rr);
  float*  lamb_re = (float*)take(sizeof(float) * nBLC * Rr);
  float*  lamb_im = (float*)take(sizeof(float) * nBLC * Rr);
  float*  gammav  = (float*)take(sizeof(float) * nBLC * Rr);
  float*  ctx     = (float*)take(sizeof(float) * nBLC);
  bf16_t* hsp_re  = (bf16_t*)take(sizeof(bf16_t) * nBLC * SP);
  bf16_t* hsp_im  = (bf16_t*)take(sizeof(bf16_t) * nBLC * SP);
  float*  hf      = (float*)take(sizeof(float) * nBLC * SP);
  bf16_t* Wmr     = (bf16_t*)take(sizeof(bf16_t) * Cc * Cc * 9);
  bf16_t* Wmi     = (bf16_t*)take(sizeof(bf16_t) * Cc * Cc * 9);
  float*  bias_m  = (float*)take(sizeof(float) * Cc);

  k_precompute_AB<<<(Cc * 64 * Rr + 255) / 256, 256, 0, stream>>>(
      U_re, U_im, V_re, V_im, Ahat_re, Ahat_im, Bhat_re, Bhat_im);
  k_merge_weights<<<(Cc * Cc * 9 + 255) / 256, 256, 0, stream>>>(
      convr_w, convr_b, convi_w, convi_b, projc_w, projc_b, Wmr, Wmi, bias_m);
  k_ctx<<<Bb * Ll * Cc, 256, 0, stream>>>(x, ctx);
  k_forcing<<<Bb * Ll, 256, 0, stream>>>(ctx, listT, plb, disp, fm_W1, fm_b1,
                                         fm_W2, fm_b2, f_scale,
                                         lamb_re, lamb_im, gammav);
  k_proj<<<Bb * Ll * 32, 256, 0, stream>>>(x, pW_re, pW_im, hs_re, hs_im);
  k_zq<<<Bb * Ll * Cc, 256, 0, stream>>>(hs_re, hs_im, Ahat_re, Ahat_im,
                                         Bhat_re, Bhat_im, zq_re, zq_im);
  k_scan<<<(Bb * Cc * Rr * Rr + 255) / 256, 256, 0, stream>>>(
      zq_re, zq_im, lamb_re, lamb_im, gammav);
  k_recon<<<Bb * Ll * Cc, 256, 0, stream>>>(zq_re, zq_im, Ahat_re, Ahat_im,
                                            Bhat_re, Bhat_im, hsp_re, hsp_im);
  k_conv<<<Bb * Ll * 16, 256, 0, stream>>>(hsp_re, hsp_im, Wmr, Wmi, bias_m, hf);
  k_ln<<<Bb * Ll * Cc, 256, 0, stream>>>(hf, x, ln_w, ln_b, out);
}
